// FusionPyramidLayer_36893769073253
// MI455X (gfx1250) — matrix operand order repaired and checked by reading, not compiled, not run on hardware
//
#include <hip/hip_runtime.h>

typedef __attribute__((ext_vector_type(16))) _Float16 v16h;
typedef __attribute__((ext_vector_type(8)))  _Float16 v8h;
typedef __attribute__((ext_vector_type(8)))  float    v8f;

namespace {
constexpr int Bn = 4;    // batch
constexpr int Cf = 64;   // feature channels
constexpr int Hs = 96, Ws = 96;
constexpr int HP = 98, WP = 98;   // halo-padded spatial dims
constexpr int OMC = 216;          // offset/mask channels (3*G*KK)
}

__device__ __forceinline__ float lrelu_f(float x) { return x >= 0.f ? x : 0.1f * x; }

__device__ __forceinline__ v8h splat8h(_Float16 x) {
  v8h v = {x, x, x, x, x, x, x, x};
  return v;
}

// ---------------------------------------------------------------------------
// Stage NCHW f32 -> halo-padded NHWC f16 (optionally scaled), channel offset co
// ---------------------------------------------------------------------------
__global__ void k_stage(const float* __restrict__ src, _Float16* __restrict__ dst,
                        int C, int Cdst, int co, float scale) {
  int e = blockIdx.x * 256 + threadIdx.x;
  int total = Bn * C * Hs * Ws;
  if (e >= total) return;
  int c = e % C; int t = e / C;
  int x = t % Ws; t /= Ws;
  int y = t % Hs; int b = t / Hs;
  float v = src[(((size_t)b * C + c) * Hs + y) * Ws + x] * scale;
  dst[(((size_t)b * HP + (y + 1)) * WP + (x + 1)) * Cdst + co + c] = (_Float16)v;
}

// ---------------------------------------------------------------------------
// Bilinear x2 upsample (align_corners=False) 48x48 -> 96x96,
// NCHW f32 -> halo-padded NHWC f16, scaled.
// ---------------------------------------------------------------------------
__global__ void k_up2(const float* __restrict__ src, _Float16* __restrict__ dst,
                      int Cdst, int co, float scale) {
  int e = blockIdx.x * 256 + threadIdx.x;
  int total = Bn * Cf * Hs * Ws;
  if (e >= total) return;
  int c = e % Cf; int t = e / Cf;
  int x = t % Ws; t /= Ws;
  int y = t % Hs; int b = t / Hs;
  float sy = fmaxf((float)y * 0.5f - 0.25f, 0.f);
  int iy0 = (int)sy; float ty = sy - (float)iy0; int iy1 = min(iy0 + 1, 47);
  float sx = fmaxf((float)x * 0.5f - 0.25f, 0.f);
  int ix0 = (int)sx; float tx = sx - (float)ix0; int ix1 = min(ix0 + 1, 47);
  const float* p = src + ((size_t)b * Cf + c) * 48 * 48;
  float v00 = p[iy0 * 48 + ix0], v01 = p[iy0 * 48 + ix1];
  float v10 = p[iy1 * 48 + ix0], v11 = p[iy1 * 48 + ix1];
  float v = (1.f - ty) * ((1.f - tx) * v00 + tx * v01) +
            ty * ((1.f - tx) * v10 + tx * v11);
  dst[(((size_t)b * HP + (y + 1)) * WP + (x + 1)) * Cdst + co + c] = (_Float16)(v * scale);
}

// ---------------------------------------------------------------------------
// Pack OIHW f32 weights into per-lane WMMA A-matrix order (f16).
// Layout: [kc][mtile][lane][16 halves], A lane mapping used here (contiguous K, same as B):
//   row = lane&15, halves 0..15 -> K=(lane>>4)*16+h.
// Logical K: mode0 (conv): kg = (ky*3+kx)*Cin + ic.
//            mode1 (dcn):  kg = (g*9+kk)*8 + cg, ic = g*8+cg.
// ---------------------------------------------------------------------------
__global__ void k_pack(const float* __restrict__ w, _Float16* __restrict__ out,
                       int Cout, int Cin, int KC, int Mtiles, int dcnMode) {
  int e = blockIdx.x * 256 + threadIdx.x;
  int total = KC * Mtiles * 512;
  if (e >= total) return;
  int h = e & 15;
  int lane = (e >> 4) & 31;
  int mt = (e >> 9) % Mtiles;
  int kc = (e >> 9) / Mtiles;
  int row = lane & 15, hi = lane >> 4, kbase = hi * 8;
  int kin = hi * 16 + h;
  int kg = kc * 32 + kin;
  int oc = mt * 16 + row;
  float val = 0.f;
  if (oc < Cout) {
    int ic, kk;
    if (dcnMode) { int g = kg / 72; int r = kg % 72; kk = r >> 3; ic = g * 8 + (r & 7); }
    else         { kk = kg / Cin; ic = kg % Cin; }
    val = w[(((size_t)oc * Cin + ic) * 3 + (kk / 3)) * 3 + (kk % 3)];
  }
  out[e] = (_Float16)val;
}

// ---------------------------------------------------------------------------
// Implicit-GEMM 3x3 conv, stride 1, pad 1 via v_wmma_f32_16x16x32_f16.
// One wave -> 64 output channels x 32 pixels. Input: halo NHWC f16.
// Outputs (any may be null):
//   out_nchw : f32 [B][64][H][W]              (always full 64 channels)
//   out_pix  : f32 [B][H][W][CoutReal]        (pixel-major, group-guarded)
//   out16    : f16 halo NHWC [B][HP][WP][Cdst16] at channel offset co16
// ---------------------------------------------------------------------------
__global__ __launch_bounds__(256) void k_conv3_wmma(
    const _Float16* __restrict__ in, int Cin,
    const _Float16* __restrict__ wpack, int KC, int Mtot,
    const float* __restrict__ bias, int CoutReal,
    float* __restrict__ out_nchw,
    float* __restrict__ out_pix,
    _Float16* __restrict__ out16, int Cdst16, int co16,
    int doLrelu) {
  int wave = blockIdx.x * 8 + (threadIdx.x >> 5);
  if (wave >= Bn * Hs * 3) return;                 // wave-uniform: EXEC stays full
  int b = wave / (Hs * 3);
  int r = wave % (Hs * 3);
  int y = r / 3;
  int x0 = (r % 3) * 32;
  int mtBase = blockIdx.y * 4;
  int ocBase = mtBase * 16;
  int lane = threadIdx.x & 31;
  int rw = lane & 15, hi = lane >> 4;

  v8f zf = {0.f, 0.f, 0.f, 0.f, 0.f, 0.f, 0.f, 0.f};
  v8f acc[8];
#pragma unroll
  for (int i = 0; i < 8; ++i) acc[i] = zf;

  for (int kc = 0; kc < KC; ++kc) {
    int kg0 = kc * 32;
    int kpos = kg0 / Cin;
    int ic0 = kg0 % Cin;
    int ky = kpos / 3, kx = kpos % 3;
    const _Float16* ap = wpack + (size_t)((kc * Mtot + mtBase) * 32 + lane) * 16;
    __builtin_prefetch(ap + (size_t)Mtot * 512, 0, 1);   // next K-chunk's A block
    v16h a0 = *(const v16h*)(ap);
    v16h a1 = *(const v16h*)(ap + 512);
    v16h a2 = *(const v16h*)(ap + 1024);
    v16h a3 = *(const v16h*)(ap + 1536);
    // B lane layout: lane<16 holds col=lane, K rows 0..15; lane>=16 K rows 16..31
    const _Float16* bp0 = in + ((size_t)(b * HP + y + ky) * WP + (x0 + kx + rw)) * Cin + ic0 + hi * 16;
    v8h b0lo = *(const v8h*)(bp0);
    v8h b0hi = *(const v8h*)(bp0 + 8);
    const _Float16* bp1 = bp0 + (size_t)16 * Cin;
    v8h b1lo = *(const v8h*)(bp1);
    v8h b1hi = *(const v8h*)(bp1 + 8);
    v16h bt0 = __builtin_shufflevector(b0lo, b0hi, 0,1,2,3,4,5,6,7,8,9,10,11,12,13,14,15);
    v16h bt1 = __builtin_shufflevector(b1lo, b1hi, 0,1,2,3,4,5,6,7,8,9,10,11,12,13,14,15);
    acc[0] = __builtin_amdgcn_wmma_f32_16x16x32_f16(false, a0, false, bt0, (short)0, acc[0], false, false);
    acc[1] = __builtin_amdgcn_wmma_f32_16x16x32_f16(false, a0, false, bt1, (short)0, acc[1], false, false);
    acc[2] = __builtin_amdgcn_wmma_f32_16x16x32_f16(false, a1, false, bt0, (short)0, acc[2], false, false);
    acc[3] = __builtin_amdgcn_wmma_f32_16x16x32_f16(false, a1, false, bt1, (short)0, acc[3], false, false);
    acc[4] = __builtin_amdgcn_wmma_f32_16x16x32_f16(false, a2, false, bt0, (short)0, acc[4], false, false);
    acc[5] = __builtin_amdgcn_wmma_f32_16x16x32_f16(false, a2, false, bt1, (short)0, acc[5], false, false);
    acc[6] = __builtin_amdgcn_wmma_f32_16x16x32_f16(false, a3, false, bt0, (short)0, acc[6], false, false);
    acc[7] = __builtin_amdgcn_wmma_f32_16x16x32_f16(false, a3, false, bt1, (short)0, acc[7], false, false);
  }

  // C/D layout: VGPR v, lane<16 -> M=v, N=lane; lane>=16 -> M=8+v, N=lane-16
#pragma unroll
  for (int mo = 0; mo < 4; ++mo) {
    int oc0 = ocBase + mo * 16 + hi * 8;
    bool gval = (oc0 + 8 <= CoutReal);    // whole 8-channel group valid (CoutReal % 8 == 0)
    float bs[8];
    if (gval) {
#pragma unroll
      for (int v = 0; v < 8; ++v) bs[v] = bias[oc0 + v];
    } else {
#pragma unroll
      for (int v = 0; v < 8; ++v) bs[v] = 0.f;
    }
#pragma unroll
    for (int nt = 0; nt < 2; ++nt) {
      int px = x0 + rw + nt * 16;
      float res[8];
#pragma unroll
      for (int v = 0; v < 8; ++v) {
        float f = acc[mo * 2 + nt][v] + bs[v];
        if (doLrelu) f = lrelu_f(f);
        res[v] = f;
      }
      if (out_nchw) {   // only used with Cout == 64: no channel guard needed
#pragma unroll
        for (int v = 0; v < 8; ++v)
          out_nchw[(((size_t)b * 64 + (oc0 + v)) * Hs + y) * Ws + px] = res[v];
      }
      if (out_pix && gval) {
        float* dp = out_pix + (((size_t)b * Hs + y) * Ws + px) * CoutReal + oc0;
        *(float4*)(dp)     = make_float4(res[0], res[1], res[2], res[3]);
        *(float4*)(dp + 4) = make_float4(res[4], res[5], res[6], res[7]);
      }
      if (out16) {
        v8h hv;
#pragma unroll
        for (int v = 0; v < 8; ++v) hv[v] = (_Float16)res[v];
        _Float16* dp = out16 + (((size_t)b * HP + (y + 1)) * WP + (px + 1)) * Cdst16 + co16 + oc0;
        *(v8h*)dp = hv;
      }
    }
  }
}

// ---------------------------------------------------------------------------
// DCN bilinear sampler: one (pixel, deformable-group, tap) -> 8 modulated
// channel values (cg 0..7) as packed f16. src = halo NHWC f16 (C stride 128).
// om = pixel-major [B][H][W][216] f32: +0 = off_y, +72 = off_x, +144 = mask.
// Blend uses packed f16 math (v_pk_fma_f16).
// ---------------------------------------------------------------------------
__device__ __forceinline__ v8h dcn_sample8(
    const _Float16* __restrict__ src, const float* __restrict__ om,
    int b, int y, int px, int pair) {
  int g = pair / 9, kk = pair % 9;
  const float* omb = om + (((size_t)b * Hs + y) * Ws + px) * OMC + (g * 9 + kk);
  float offy = omb[0];
  float offx = omb[72];
  float mv   = omb[144];
  float mask = 1.f / (1.f + __expf(-mv));
  float py  = offy + (float)(y + kk / 3 - 1);
  float pxf = offx + (float)(px + kk % 3 - 1);
  float y0f = floorf(py), x0f = floorf(pxf);
  float ty = py - y0f, tx = pxf - x0f;
  int iy0 = (int)y0f, ix0 = (int)x0f;
  int iy1 = iy0 + 1, ix1 = ix0 + 1;
  float w00 = (1.f - ty) * (1.f - tx), w01 = (1.f - ty) * tx;
  float w10 = ty * (1.f - tx),         w11 = ty * tx;
  if (iy0 < 0 || iy0 >= Hs) { w00 = 0.f; w01 = 0.f; }
  if (iy1 < 0 || iy1 >= Hs) { w10 = 0.f; w11 = 0.f; }
  if (ix0 < 0 || ix0 >= Ws) { w00 = 0.f; w10 = 0.f; }
  if (ix1 < 0 || ix1 >= Ws) { w01 = 0.f; w11 = 0.f; }
  int cy0 = min(max(iy0, 0), Hs - 1), cy1 = min(max(iy1, 0), Hs - 1);
  int cx0 = min(max(ix0, 0), Ws - 1), cx1 = min(max(ix1, 0), Ws - 1);
  int cb = g * 8;
  const _Float16* p00 = src + (((size_t)b * HP + (cy0 + 1)) * WP + (cx0 + 1)) * 128 + cb;
  const _Float16* p01 = src + (((size_t)b * HP + (cy0 + 1)) * WP + (cx1 + 1)) * 128 + cb;
  const _Float16* p10 = src + (((size_t)b * HP + (cy1 + 1)) * WP + (cx0 + 1)) * 128 + cb;
  const _Float16* p11 = src + (((size_t)b * HP + (cy1 + 1)) * WP + (cx1 + 1)) * 128 + cb;
  v8h h00 = *(const v8h*)p00;
  v8h h01 = *(const v8h*)p01;
  v8h h10 = *(const v8h*)p10;
  v8h h11 = *(const v8h*)p11;
  v8h r = h00 * splat8h((_Float16)(w00 * mask)) +
          h01 * splat8h((_Float16)(w01 * mask)) +
          h10 * splat8h((_Float16)(w10 * mask)) +
          h11 * splat8h((_Float16)(w11 * mask));
  return r;
}

// ---------------------------------------------------------------------------
// Fused modulated deformable conv: sampling + K=576 GEMM via WMMA.
// Output (f16, + bias, no activation) -> catF channels 0..63.
// ---------------------------------------------------------------------------
__global__ __launch_bounds__(256) void k_dcn_wmma(
    const _Float16* __restrict__ src, const float* __restrict__ om,
    const _Float16* __restrict__ wpack, const float* __restrict__ bias,
    _Float16* __restrict__ out16) {
  int wave = blockIdx.x * 8 + (threadIdx.x >> 5);
  if (wave >= Bn * Hs * 3) return;
  int b = wave / (Hs * 3);
  int r = wave % (Hs * 3);
  int y = r / 3;
  int x0 = (r % 3) * 32;
  int lane = threadIdx.x & 31;
  int rw = lane & 15, hi = lane >> 4;

  v8f zf = {0.f, 0.f, 0.f, 0.f, 0.f, 0.f, 0.f, 0.f};
  v8f acc[8];
#pragma unroll
  for (int i = 0; i < 8; ++i) acc[i] = zf;

  for (int kc = 0; kc < 18; ++kc) {
    const _Float16* ap = wpack + (size_t)(kc * 4 * 32 + lane) * 16;
    __builtin_prefetch(ap + 2048, 0, 1);
    v16h a0 = *(const v16h*)(ap);
    v16h a1 = *(const v16h*)(ap + 512);
    v16h a2 = *(const v16h*)(ap + 1024);
    v16h a3 = *(const v16h*)(ap + 1536);
    v16h bts[2];
#pragma unroll
    for (int nt = 0; nt < 2; ++nt) {
      int px = x0 + rw + nt * 16;
      // lane's B rows: K = hi*16 + h, halves 0..7 -> pair 2*hi, halves 8..15 -> pair 2*hi+1
      v8h s0 = dcn_sample8(src, om, b, y, px, kc * 4 + hi * 2);
      v8h s1 = dcn_sample8(src, om, b, y, px, kc * 4 + hi * 2 + 1);
      bts[nt] = __builtin_shufflevector(s0, s1, 0,1,2,3,4,5,6,7,8,9,10,11,12,13,14,15);
    }
    acc[0] = __builtin_amdgcn_wmma_f32_16x16x32_f16(false, a0, false, bts[0], (short)0, acc[0], false, false);
    acc[1] = __builtin_amdgcn_wmma_f32_16x16x32_f16(false, a0, false, bts[1], (short)0, acc[1], false, false);
    acc[2] = __builtin_amdgcn_wmma_f32_16x16x32_f16(false, a1, false, bts[0], (short)0, acc[2], false, false);
    acc[3] = __builtin_amdgcn_wmma_f32_16x16x32_f16(false, a1, false, bts[1], (short)0, acc[3], false, false);
    acc[4] = __builtin_amdgcn_wmma_f32_16x16x32_f16(false, a2, false, bts[0], (short)0, acc[4], false, false);
    acc[5] = __builtin_amdgcn_wmma_f32_16x16x32_f16(false, a2, false, bts[1], (short)0, acc[5], false, false);
    acc[6] = __builtin_amdgcn_wmma_f32_16x16x32_f16(false, a3, false, bts[0], (short)0, acc[6], false, false);
    acc[7] = __builtin_amdgcn_wmma_f32_16x16x32_f16(false, a3, false, bts[1], (short)0, acc[7], false, false);
  }

#pragma unroll
  for (int mo = 0; mo < 4; ++mo) {
    int oc0 = mo * 16 + hi * 8;
    float bs[8];
#pragma unroll
    for (int v = 0; v < 8; ++v) bs[v] = bias[oc0 + v];
#pragma unroll
    for (int nt = 0; nt < 2; ++nt) {
      int px = x0 + rw + nt * 16;
      v8h hv;
#pragma unroll
      for (int v = 0; v < 8; ++v) hv[v] = (_Float16)(acc[mo * 2 + nt][v] + bs[v]);
      _Float16* dp = out16 + (((size_t)b * HP + (y + 1)) * WP + (px + 1)) * 128 + oc0;
      *(v8h*)dp = hv;
    }
  }
}

// ---------------------------------------------------------------------------
// Host orchestration
// ---------------------------------------------------------------------------
extern "C" void kernel_launch(void* const* d_in, const int* in_sizes, int n_in,
                              void* d_out, int out_size, void* d_ws, size_t ws_size,
                              hipStream_t stream) {
  const float* src0         = (const float*)d_in[0];
  const float* src1         = (const float*)d_in[1];
  const float* last_offset  = (const float*)d_in[2];
  const float* last_feature = (const float*)d_in[3];
  const float* w_off1 = (const float*)d_in[4];
  const float* b_off1 = (const float*)d_in[5];
  const float* w_off2 = (const float*)d_in[6];
  const float* b_off2 = (const float*)d_in[7];
  const float* w_off3 = (const float*)d_in[8];
  const float* b_off3 = (const float*)d_in[9];
  const float* w_om   = (const float*)d_in[10];
  const float* b_om   = (const float*)d_in[11];
  const float* w_dcn  = (const float*)d_in[12];
  const float* b_dcn  = (const float*)d_in[13];
  const float* w_fea  = (const float*)d_in[14];
  const float* b_fea  = (const float*)d_in[15];

  char* wsp = (char*)d_ws;
  size_t off = 0;
  auto wsAlloc = [&](size_t bytes) -> void* {
    void* p = wsp + off;
    off += (bytes + 255) & ~(size_t)255;
    return p;
  };
  const size_t SZ_CAT = (size_t)Bn * HP * WP * 128 * sizeof(_Float16);
  const size_t SZ_64  = (size_t)Bn * HP * WP * 64  * sizeof(_Float16);
  const size_t SZ_OM  = (size_t)Bn * Hs * Ws * OMC * sizeof(float);
  _Float16* cat1 = (_Float16*)wsAlloc(SZ_CAT);   // [src0 | src1] halo NHWC f16
  _Float16* cat2 = (_Float16*)wsAlloc(SZ_CAT);   // [conv1 out | 2*up2(last_offset)]
  _Float16* catF = (_Float16*)wsAlloc(SZ_CAT);   // [dcn out | up2(last_feature)]
  _Float16* off2 = (_Float16*)wsAlloc(SZ_64);    // conv2 out (halo NHWC)
  _Float16* off3 = (_Float16*)wsAlloc(SZ_64);    // conv3 out (halo NHWC)
  float*    omb  = (float*)wsAlloc(SZ_OM);       // offset/mask conv out, pixel-major
  _Float16* wp1  = (_Float16*)wsAlloc((size_t)36 * 4  * 512 * 2);
  _Float16* wp2  = (_Float16*)wsAlloc((size_t)36 * 4  * 512 * 2);
  _Float16* wp3  = (_Float16*)wsAlloc((size_t)18 * 4  * 512 * 2);
  _Float16* wpom = (_Float16*)wsAlloc((size_t)18 * 16 * 512 * 2);
  _Float16* wpd  = (_Float16*)wsAlloc((size_t)18 * 4  * 512 * 2);
  _Float16* wpf  = (_Float16*)wsAlloc((size_t)36 * 4  * 512 * 2);

  // Zero halo-padded staging buffers (capture-legal async memset)
  (void)hipMemsetAsync(cat1, 0, SZ_CAT, stream);
  (void)hipMemsetAsync(cat2, 0, SZ_CAT, stream);
  (void)hipMemsetAsync(catF, 0, SZ_CAT, stream);
  (void)hipMemsetAsync(off2, 0, SZ_64, stream);
  (void)hipMemsetAsync(off3, 0, SZ_64, stream);

  // Weight packing
  auto pack = [&](const float* w, _Float16* o, int Cout, int Cin, int KC, int Mt, int dcn) {
    int total = KC * Mt * 512;
    k_pack<<<(total + 255) / 256, 256, 0, stream>>>(w, o, Cout, Cin, KC, Mt, dcn);
  };
  pack(w_off1, wp1, 64, 128, 36, 4, 0);
  pack(w_off2, wp2, 64, 128, 36, 4, 0);
  pack(w_off3, wp3, 64, 64, 18, 4, 0);
  pack(w_om, wpom, OMC, 64, 18, 16, 0);
  pack(w_dcn, wpd, 64, 64, 18, 4, 1);
  pack(w_fea, wpf, 64, 128, 36, 4, 0);

  // Activation staging + upsamples
  int nst = Bn * 64 * Hs * Ws;
  int gst = (nst + 255) / 256;
  k_stage<<<gst, 256, 0, stream>>>(src0, cat1, 64, 128, 0, 1.f);
  k_stage<<<gst, 256, 0, stream>>>(src1, cat1, 64, 128, 64, 1.f);
  k_up2<<<gst, 256, 0, stream>>>(last_offset, cat2, 128, 64, 2.f);
  k_up2<<<gst, 256, 0, stream>>>(last_feature, catF, 128, 64, 1.f);

  float* out_offset  = (float*)d_out;
  float* out_feature = out_offset + (size_t)Bn * 64 * Hs * Ws;

  dim3 blk(256);
  dim3 g1(144, 1);   // 1152 waves = B*H*(W/32) tiles
  dim3 g4(144, 4);   // 4 oc-groups of 64 for the 216-channel om conv

  // conv1: concat(src0,src1) -> offset1 (lrelu) -> cat2[0..63]
  k_conv3_wmma<<<g1, blk, 0, stream>>>(cat1, 128, wp1, 36, 4, b_off1, 64,
                                       nullptr, nullptr, cat2, 128, 0, 1);
  // conv2: concat(offset1, 2*lo) -> offset2 (lrelu) -> off2
  k_conv3_wmma<<<g1, blk, 0, stream>>>(cat2, 128, wp2, 36, 4, b_off2, 64,
                                       nullptr, nullptr, off2, 64, 0, 1);
  // conv3: offset2 -> offset (lrelu) -> d_out[0] (f32 NCHW) + off3 (f16)
  k_conv3_wmma<<<g1, blk, 0, stream>>>(off2, 64, wp3, 18, 4, b_off3, 64,
                                       out_offset, nullptr, off3, 64, 0, 1);
  // om conv: offset -> [off_y|off_x|mask] (no activation) -> omb (pixel-major f32)
  k_conv3_wmma<<<g4, blk, 0, stream>>>(off3, 64, wpom, 18, 16, b_om, OMC,
                                       nullptr, omb, nullptr, 0, 0, 0);
  // fused DCNv2: sample src0 (cat1 ch0..63) with omb, GEMM w_dcn -> catF[0..63]
  k_dcn_wmma<<<g1, blk, 0, stream>>>(cat1, omb, wpd, b_dcn, catF);
  // fea conv: concat(dcn_feature, lf) -> feature (lrelu) -> d_out[1]
  k_conv3_wmma<<<g1, blk, 0, stream>>>(catF, 128, wpf, 36, 4, b_fea, 64,
                                       out_feature, nullptr, nullptr, 0, 0, 1);
}
